// LinearCDE_3246995275973
// MI455X (gfx1250) — matrix operand order repaired and checked by reading, not compiled, not run on hardware
//
#include <hip/hip_runtime.h>
#include <hip/hip_bf16.h>
#include <stdint.h>

// Problem constants (B,S,D,H) = (32,1024,32,64)
#define B_    32
#define S_    1024
#define T_    1023          // scan steps
#define D_    32
#define H_    64
#define KIN   33            // D+1 feature dim
#define NA    4096          // H*H columns for W_A
#define NTOT  4160          // + H columns for W_B
#define NCHUNK 65           // N tiles of 64 cols
#define LDSTR 66            // bf16 elems per row (odd dword count -> conflict free)
#define LDSTRU 33           // same stride in dwords
#define WCHB  (64 * LDSTR * 2)   // bytes per W chunk image = 8448

typedef __attribute__((ext_vector_type(16))) __bf16 v16bf;
typedef __attribute__((ext_vector_type(8)))  float  v8f;

__device__ __forceinline__ unsigned short f2bf(float f) {
    unsigned int u = __float_as_uint(f);
    u += 0x7FFFu + ((u >> 16) & 1u);          // round-to-nearest-even
    return (unsigned short)(u >> 16);
}
__device__ __forceinline__ unsigned int pk2bf(float lo, float hi) {   // packed RNE cvt
    __hip_bfloat162 p = __float22bfloat162_rn(float2{lo, hi});
    union { __hip_bfloat162 h; unsigned int u; } cv;
    cv.h = p;
    return cv.u;
}
__device__ __forceinline__ float bflo2f(unsigned int u) { return __uint_as_float(u << 16); }
__device__ __forceinline__ float bfhi2f(unsigned int u) { return __uint_as_float(u & 0xFFFF0000u); }

// ---------------------------------------------------------------------------
// Kernel 0: pre-convert Wcat=[W_A;W_B] to bf16, padded to LDSTR (zeros K 33..65).
// ---------------------------------------------------------------------------
__global__ __launch_bounds__(256)
void cde_wpack(const float* __restrict__ W_A, const float* __restrict__ W_B,
               unsigned short* __restrict__ Wp)
{
    const int e = blockIdx.x * 256 + threadIdx.x;
    if (e >= NTOT * LDSTR) return;
    const int c = e / LDSTR;
    const int k = e - c * LDSTR;
    float v = 0.0f;
    if (k < KIN) v = (c < NA) ? W_A[(size_t)c * KIN + k] : W_B[(size_t)(c - NA) * KIN + k];
    Wp[e] = f2bf(v);
}

// ---------------------------------------------------------------------------
// Kernel 1: feature GEMM. One WG per 128-row M-tile; N loop inside; A-fragments
// register-resident; W chunks async-copied into a 3-slot LDS ring (ASYNCcnt);
// branch-free epilogue: packed bf16 converts + precomputed row offsets + dump.
// ---------------------------------------------------------------------------
__global__ __launch_bounds__(256)
void cde_feature_gemm(const float* __restrict__ X,
                      const unsigned short* __restrict__ Wp,
                      unsigned short* __restrict__ As,
                      float* __restrict__ Bs,
                      unsigned int dumpA,      // ushort-offset from As to dump
                      unsigned int dumpB)      // float-offset from Bs to dump
{
    __shared__ unsigned short Ash[128 * LDSTR];      // 16.9 KB (input staging)
    __shared__ unsigned short Wq[3][64 * LDSTR];     // 3 x 8.4 KB async ring

    const int tid   = threadIdx.x;
    const int tileM = blockIdx.x;   // 0..255

    // ---- stage A rows ([ts, X]) to LDS as bf16, zero-pad K 33..65 ----
    {
        const int row  = tid >> 1;
        const int half = tid & 1;
        const int gr = tileM * 128 + row;   // b*1024 + t
        const float* xp = X + (size_t)gr * D_ + half * 16;
        unsigned short* dst = Ash + row * LDSTR + 1 + half * 16;
        #pragma unroll
        for (int i = 0; i < 16; ++i) dst[i] = f2bf(xp[i]);
        if (half == 0) {
            Ash[row * LDSTR] = f2bf(1.0f / (float)T_);
        } else {
            unsigned short* pz = Ash + row * LDSTR + KIN;
            #pragma unroll
            for (int i = 0; i < LDSTR - KIN; ++i) pz[i] = 0;
        }
    }

    // ---- async W ring ----
    const unsigned int wqBase = (unsigned int)(uintptr_t)&Wq[0][0];
    auto issueW = [&](int ch) {                       // copy 8448B chunk -> slot ch%3
        const unsigned int lb = wqBase + (unsigned)(ch % 3) * WCHB + (unsigned)tid * 16u;
        const unsigned long long g =
            (unsigned long long)(uintptr_t)((const unsigned char*)Wp + (size_t)ch * WCHB) +
            (unsigned long long)tid * 16ull;
        asm volatile("global_load_async_to_lds_b128 %0, %1, off" :: "v"(lb), "v"(g) : "memory");
        asm volatile("global_load_async_to_lds_b128 %0, %1, off" :: "v"(lb + 4096u), "v"(g + 4096ull) : "memory");
        if (tid < 16)   // tail 256B (wave0, partial EXEC)
            asm volatile("global_load_async_to_lds_b128 %0, %1, off" :: "v"(lb + 8192u), "v"(g + 8192ull) : "memory");
    };
    issueW(0);
    issueW(1);
    __syncthreads();            // Ash staged

    const int lane = tid & 31;
    const int w    = tid >> 5;
    const int wm   = w >> 1;        // 0..3
    const int wn   = w & 1;         // 0..1
    const int l15  = lane & 15;
    const int lhi  = lane >> 4;

    union Frag { v16bf v; unsigned int u[8]; };

    // ---- A fragments: loaded ONCE, reused for all 65 N-chunks ----
    const unsigned int* AshU = (const unsigned int*)Ash;
    Frag afrag[2][2];   // [mi][ks]
    #pragma unroll
    for (int ks = 0; ks < 2; ++ks)
        #pragma unroll
        for (int mi = 0; mi < 2; ++mi) {
            const int row = wm * 32 + mi * 16 + l15;
            const int kb  = ks * 32 + lhi * 8;        // A layout: lane-half +8 K split
            const unsigned int* p = AshU + row * LDSTRU + (kb >> 1);
            #pragma unroll
            for (int v = 0; v < 4; ++v) { afrag[mi][ks].u[v] = p[v]; afrag[mi][ks].u[4 + v] = p[8 + v]; }
        }

    // ---- precomputed branch-free store offsets ----
    const int rowbase = tileM * 128 + wm * 32;
    unsigned int rowOffA[2][8];
    #pragma unroll
    for (int mi = 0; mi < 2; ++mi)
        #pragma unroll
        for (int v = 0; v < 8; ++v) {
            const int gr = rowbase + mi * 16 + v + lhi * 8;
            const int b = gr >> 10;
            const int t = gr & 1023;
            rowOffA[mi][v] = t ? (unsigned int)((b * T_ + (t - 1)) * NA) : dumpA;
        }
    const unsigned int colLane0 = (unsigned)(wn * 32 + l15);

    for (int ch = 0; ch < NCHUNK; ++ch) {
        if (tid < 32) asm volatile("s_wait_asynccnt 3" ::: "memory");  // own slot(ch) done
        else          asm volatile("s_wait_asynccnt 2" ::: "memory");
        __syncthreads();                 // all waves' slot(ch) ready; slot(ch+2) free
        if (ch + 2 < NCHUNK) issueW(ch + 2);

        const unsigned int* WshU = (const unsigned int*)&Wq[ch % 3][0];

        v8f acc[2][2] = {};
        #pragma unroll
        for (int ks = 0; ks < 2; ++ks) {
            Frag bf[2];                  // B layout: lane-half +8 K split
            #pragma unroll
            for (int ni = 0; ni < 2; ++ni) {
                const int col = wn * 32 + ni * 16 + l15;
                const int kb  = ks * 32 + lhi * 8;
                const unsigned int* p = WshU + col * LDSTRU + (kb >> 1);
                #pragma unroll
                for (int v = 0; v < 4; ++v) { bf[ni].u[v] = p[v]; bf[ni].u[4 + v] = p[8 + v]; }
            }
            #pragma unroll
            for (int mi = 0; mi < 2; ++mi)
                #pragma unroll
                for (int ni = 0; ni < 2; ++ni)
                    acc[mi][ni] = __builtin_amdgcn_wmma_f32_16x16x32_bf16(
                        false, afrag[mi][ks].v, false, bf[ni].v,
                        (short)0, acc[mi][ni], false, false);
        }

        if (ch < 64) {
            // bf16 As stores (packed converts; each row-chunk = one 128B L2 line)
            const unsigned int chcol = (unsigned)(ch * 64) + colLane0;
            #pragma unroll
            for (int mi = 0; mi < 2; ++mi)
                #pragma unroll
                for (int ni = 0; ni < 2; ++ni) {
                    const unsigned int cc = chcol + (unsigned)(ni * 16);
                    #pragma unroll
                    for (int v = 0; v < 8; v += 2) {
                        const unsigned int pu = pk2bf(acc[mi][ni][v], acc[mi][ni][v + 1]);
                        As[rowOffA[mi][v]     + cc] = (unsigned short)(pu & 0xFFFFu);
                        As[rowOffA[mi][v + 1] + cc] = (unsigned short)(pu >> 16);
                    }
                }
        } else {
            // W_B columns -> Bs (f32)
            #pragma unroll
            for (int mi = 0; mi < 2; ++mi)
                #pragma unroll
                for (int v = 0; v < 8; ++v) {
                    const int gr = rowbase + mi * 16 + v + lhi * 8;
                    const int b = gr >> 10;
                    const int t = gr & 1023;
                    const unsigned int ro = t ? (unsigned int)((b * T_ + (t - 1)) * H_) : dumpB;
                    #pragma unroll
                    for (int ni = 0; ni < 2; ++ni)
                        Bs[ro + colLane0 + (unsigned)(ni * 16)] = acc[mi][ni][v];
                }
        }
    }
}

// ---------------------------------------------------------------------------
// Kernel 2: per-batch sequential scan  y_t = y_{t-1} + A_t y_{t-1} + B_t.
// 16-stage async LDS ring (depth-15 HBM prefetch); 64 rows x 4 chunks; 4-way
// FMA ILP; shfl_xor reduce; one barrier per step.
// ---------------------------------------------------------------------------
#define RSTG 16
#define STGB (8192 + 256 + 256)         // A(8KB) + B(256B) + dummy(256B)

__global__ __launch_bounds__(256)
void cde_scan(const float* __restrict__ X,
              const float* __restrict__ W_init,
              const float* __restrict__ b_init,
              const unsigned short* __restrict__ As,
              const float* __restrict__ Bs,
              float* __restrict__ out)
{
    __shared__ __align__(16) unsigned char ring[RSTG * STGB];   // ~139 KB
    __shared__ __align__(16) float ybuf[2][H_];

    const int tid = threadIdx.x;
    const int bb  = blockIdx.x;
    const int i   = tid >> 2;   // output row 0..63
    const int c   = tid & 3;    // 16-wide k-chunk

    if (tid < H_) {             // y0 = X[b,0,:] @ W_init.T + b_init
        const float* xp = X + (size_t)bb * S_ * D_;
        const float* wp = W_init + tid * D_;
        float acc = b_init[tid];
        #pragma unroll
        for (int d = 0; d < D_; ++d) acc = fmaf(xp[d], wp[d], acc);
        ybuf[0][tid] = acc;
        out[(size_t)bb * S_ * H_ + tid] = acc;
    }

    const unsigned short* Ab = As + (size_t)bb * T_ * (H_ * H_);
    const float*          Bb = Bs + (size_t)bb * T_ * H_;
    const unsigned int ringBase = (unsigned int)(uintptr_t)&ring[0];

    auto issueStage = [&](int tt) {     // stage for step tt -> slot tt%RSTG
        const unsigned int sbase = ringBase + (unsigned)(tt % RSTG) * STGB;
        const unsigned long long gA =
            (unsigned long long)(uintptr_t)((const unsigned char*)Ab + (size_t)(tt - 1) * 8192 + tid * 32);
        const unsigned int lA = sbase + (unsigned)tid * 32;
        asm volatile("global_load_async_to_lds_b128 %0, %1, off" :: "v"(lA), "v"(gA) : "memory");
        asm volatile("global_load_async_to_lds_b128 %0, %1, off" :: "v"(lA + 16), "v"(gA + 16) : "memory");
        if (tid < 32) {                 // wave0 fetches B_t (lanes 16..31 -> dummy)
            const int l16 = tid & 15;
            const unsigned long long gB =
                (unsigned long long)(uintptr_t)((const unsigned char*)(Bb + (size_t)(tt - 1) * H_) + l16 * 16);
            const unsigned int lB = sbase + 8192u + ((tid >> 4) ? 256u : 0u) + (unsigned)l16 * 16;
            asm volatile("global_load_async_to_lds_b128 %0, %1, off" :: "v"(lB), "v"(gB) : "memory");
        }
    };
    auto waitRing = [&]() {             // oldest needed stage complete (own issues)
        if (tid < 32) asm volatile("s_wait_asynccnt 42" ::: "memory");  // 3*(RSTG-2)
        else          asm volatile("s_wait_asynccnt 28" ::: "memory");  // 2*(RSTG-2)
    };

    for (int tt = 1; tt <= RSTG - 1; ++tt) issueStage(tt);
    waitRing();
    __syncthreads();                    // stage(1) globally ready; y0 published

    int pp = 0;
    for (int t = 1; t <= T_; ++t) {
        if (t + RSTG - 1 <= T_) issueStage(t + RSTG - 1);  // refill slot freed last step

        const unsigned char* stg = &ring[(unsigned)(t % RSTG) * STGB];
        const uint4* ap = (const uint4*)(stg + (unsigned)tid * 32);
        const uint4 a0 = ap[0], a1 = ap[1];

        const float4* yv4 = (const float4*)&ybuf[pp][c * 16];
        const float4 y0 = yv4[0], y1 = yv4[1], y2 = yv4[2], y3 = yv4[3];

        float s0 = 0.0f, s1 = 0.0f, s2 = 0.0f, s3 = 0.0f;   // 4-way ILP
        s0 = fmaf(bflo2f(a0.x), y0.x, s0); s0 = fmaf(bfhi2f(a0.x), y0.y, s0);
        s1 = fmaf(bflo2f(a0.y), y0.z, s1); s1 = fmaf(bfhi2f(a0.y), y0.w, s1);
        s2 = fmaf(bflo2f(a0.z), y1.x, s2); s2 = fmaf(bfhi2f(a0.z), y1.y, s2);
        s3 = fmaf(bflo2f(a0.w), y1.z, s3); s3 = fmaf(bfhi2f(a0.w), y1.w, s3);
        s0 = fmaf(bflo2f(a1.x), y2.x, s0); s0 = fmaf(bfhi2f(a1.x), y2.y, s0);
        s1 = fmaf(bflo2f(a1.y), y2.z, s1); s1 = fmaf(bfhi2f(a1.y), y2.w, s1);
        s2 = fmaf(bflo2f(a1.z), y3.x, s2); s2 = fmaf(bfhi2f(a1.z), y3.y, s2);
        s3 = fmaf(bflo2f(a1.w), y3.z, s3); s3 = fmaf(bfhi2f(a1.w), y3.w, s3);
        float part = (s0 + s1) + (s2 + s3);

        part += __shfl_xor(part, 1, 32);
        part += __shfl_xor(part, 2, 32);

        if (c == 0) {
            const float bt = ((const float*)(stg + 8192))[i];
            const float yn = ybuf[pp][i] + part + bt;
            ybuf[pp ^ 1][i] = yn;
            out[((size_t)bb * S_ + t) * H_ + i] = yn;
        }
        waitRing();
        __syncthreads();                // publish y_t; slot t free for reuse
        pp ^= 1;
    }
}

// ---------------------------------------------------------------------------
extern "C" void kernel_launch(void* const* d_in, const int* in_sizes, int n_in,
                              void* d_out, int out_size, void* d_ws, size_t ws_size,
                              hipStream_t stream) {
    const float* X      = (const float*)d_in[0];
    const float* W_init = (const float*)d_in[1];
    const float* b_init = (const float*)d_in[2];
    const float* W_A    = (const float*)d_in[3];
    const float* W_B    = (const float*)d_in[4];
    float* out = (float*)d_out;

    const size_t asBytes = (size_t)B_ * T_ * NA * sizeof(unsigned short); // 268 MB
    const size_t bsBytes = (size_t)B_ * T_ * H_ * sizeof(float);          // 8.4 MB
    const size_t wpBytes = (size_t)NTOT * LDSTR * sizeof(unsigned short); // 549 KB

    unsigned short* As = (unsigned short*)d_ws;
    float*          Bs = (float*)((char*)d_ws + asBytes);
    unsigned short* Wp = (unsigned short*)((char*)d_ws + asBytes + bsBytes);
    const unsigned int dumpA = (unsigned int)((asBytes + bsBytes + wpBytes) / 2); // ushort off from As
    const unsigned int dumpB = (unsigned int)((bsBytes + wpBytes) / 4);           // float off from Bs

    cde_wpack<<<dim3((NTOT * LDSTR + 255) / 256), dim3(256), 0, stream>>>(W_A, W_B, Wp);
    cde_feature_gemm<<<dim3((B_ * S_) / 128), dim3(256), 0, stream>>>(X, Wp, As, Bs, dumpA, dumpB);
    cde_scan<<<dim3(B_), dim3(256), 0, stream>>>(X, W_init, b_init, As, Bs, out);
}
